// MVSTFN_submodule_42863773614350
// MI455X (gfx1250) — matrix operand order repaired and checked by reading, not compiled, not run on hardware
//
#include <hip/hip_runtime.h>
#include <math.h>

typedef __attribute__((ext_vector_type(16))) _Float16 v16h;
typedef __attribute__((ext_vector_type(8)))  float    v8f;

#define NEG_SLOPE 0.01f
#define LN_EPS 1e-5f
#define NORM_EPS 1e-12f

__device__ __forceinline__ float leaky(float v) { return v >= 0.f ? v : NEG_SLOPE * v; }

// K index for element e (0..15) of a v16h A-fragment (contiguous order, matches B):
// lanes 0-15 : K=0..15
// lanes 16-31: K=16..31
__device__ __forceinline__ int a_kidx(int lane, int e) {
    return ((lane >> 4) << 4) + e;
}

// ---------------------------------------------------------------------------
// Kernel 1: time conv  nf[b,n,f] = sum_t x[b,n,t,f]*tcw[t] + tcb
// ---------------------------------------------------------------------------
__global__ __launch_bounds__(256)
void nf_kernel(const float* __restrict__ x, const float* __restrict__ tcw,
               const float* __restrict__ tcb, float* __restrict__ nf) {
    const int T = 12, F = 32, TOT = 2 * 1024 * 32;
    int idx = blockIdx.x * blockDim.x + threadIdx.x;
    if (idx >= TOT) return;
    int f = idx & 31;
    int bn = idx >> 5;
    float acc = tcb[0];
#pragma unroll
    for (int t = 0; t < T; ++t) acc += x[((size_t)bn * T + t) * F + f] * tcw[t];
    nf[idx] = acc;
}

// ---------------------------------------------------------------------------
// Kernel 2 (WMMA): fused pairwise sim-network.
// One workgroup (4 waves) per (b,i); each wave loops over 16-wide j tiles.
// sim[b,i,j] = sigmoid(wout . leaky(W1 . leaky(W0 . |nf_i - nf_j|)) + bout)
// ---------------------------------------------------------------------------
__global__ __launch_bounds__(128)
void sim_kernel(const float* __restrict__ nf, const float* __restrict__ w0,
                const float* __restrict__ w1, const float* __restrict__ wout,
                const float* __restrict__ bout_p, float* __restrict__ sim) {
    const int N = 1024, F = 32;
    __shared__ _Float16 s_w0[64 * 32];       // (2E=64, F=32)
    __shared__ _Float16 s_w1[32 * 64];       // (E=32, 2E=64)
    __shared__ float    s_wout[32];
    __shared__ _Float16 s_xij[4][16 * 32];   // |nf_j - nf_i| tile, f16
    __shared__ _Float16 s_h  [4][16 * 64];
    __shared__ float    s_h2 [4][16 * 32];

    const int tid = threadIdx.x, wave = tid >> 5, lane = tid & 31;
    const int bg = blockIdx.x, b = bg / N, i = bg % N;

    for (int k = tid; k < 64 * 32; k += 128) s_w0[k] = (_Float16)w0[k];
    for (int k = tid; k < 32 * 64; k += 128) s_w1[k] = (_Float16)w1[k];
    if (tid < 32) s_wout[tid] = wout[tid];
    __syncthreads();
    const float bout = bout_p[0];

    // nf_i feature owned by this lane (lane == feature index, F == 32)
    const float nfi_val = nf[((size_t)b * N + i) * F + lane];

    // Per-wave resident B fragments (32x16 f16 B layout: n=lane&15, K=(lane>=16?16:0)+e)
    const int ncol = lane & 15;
    const int khalf = (lane >> 4) << 4;
    v16h b0[4];
#pragma unroll
    for (int nt = 0; nt < 4; ++nt)
#pragma unroll
        for (int e = 0; e < 16; ++e)
            b0[nt][e] = s_w0[(nt * 16 + ncol) * 32 + khalf + e];
    v16h b1[2][2];
#pragma unroll
    for (int nt = 0; nt < 2; ++nt)
#pragma unroll
        for (int kk = 0; kk < 2; ++kk)
#pragma unroll
            for (int e = 0; e < 16; ++e)
                b1[nt][kk][e] = s_w1[(nt * 16 + ncol) * 64 + kk * 32 + khalf + e];

    const int m = lane & 15;
    const int rowbase = (lane >> 4) << 3;

    for (int jt = wave; jt < N / 16; jt += 4) {   // uniform 16 iterations per wave
        const int j0 = jt * 16;
        // |nf_j - nf_i| -> LDS (lane-strided: no adjacent fp ops, scalar fabs only)
#pragma unroll
        for (int r = 0; r < 16; ++r) {
            float v = nf[((size_t)b * N + j0 + r) * F + lane];
            s_xij[wave][r * 32 + lane] = (_Float16)fabsf(v - nfi_val);
        }
        __syncthreads();

        // A fragment: pure f16 LDS loads, no math
        v16h a;
#pragma unroll
        for (int e = 0; e < 16; ++e)
            a[e] = s_xij[wave][m * 32 + a_kidx(lane, e)];

        // GEMM1: (16x32) x (32x64) -> h, leaky, stage to LDS as f16
#pragma unroll
        for (int nt = 0; nt < 4; ++nt) {
            v8f c = {};
            c = __builtin_amdgcn_wmma_f32_16x16x32_f16(false, a, false, b0[nt],
                                                       (short)0, c, false, false);
#pragma unroll
            for (int r = 0; r < 8; ++r)
                s_h[wave][(rowbase + r) * 64 + nt * 16 + ncol] = (_Float16)leaky(c[r]);
        }
        __syncthreads();
        // GEMM2: (16x64) x (64x32) -> h2, leaky
#pragma unroll
        for (int nt = 0; nt < 2; ++nt) {
            v8f c2 = {};
#pragma unroll
            for (int kk = 0; kk < 2; ++kk) {
                v16h a2;
#pragma unroll
                for (int e = 0; e < 16; ++e)
                    a2[e] = s_h[wave][m * 64 + kk * 32 + a_kidx(lane, e)];
                c2 = __builtin_amdgcn_wmma_f32_16x16x32_f16(false, a2, false, b1[nt][kk],
                                                            (short)0, c2, false, false);
            }
#pragma unroll
            for (int r = 0; r < 8; ++r)
                s_h2[wave][(rowbase + r) * 32 + nt * 16 + ncol] = leaky(c2[r]);
        }
        __syncthreads();
        // sim = sigmoid(h2 . wout + bout); two half-lanes per row, combine via shfl
        {
            float p = 0.f;
            const int dbase = (lane >> 4) << 4;
#pragma unroll
            for (int d = 0; d < 16; ++d)
                p += s_h2[wave][m * 32 + dbase + d] * s_wout[dbase + d];
            p += __shfl_xor(p, 16, 32);
            if (lane < 16)
                sim[((size_t)b * N + i) * N + j0 + m] = 1.f / (1.f + __expf(-(p + bout)));
        }
        __syncthreads();
    }
}

// ---------------------------------------------------------------------------
// Kernel 3: per-row sums  merge_sum = sum_j edge*adj ; ssum = sum_j |sim*edge*adj|
// ---------------------------------------------------------------------------
__global__ __launch_bounds__(256)
void rowsum_kernel(const float* __restrict__ sim, const float* __restrict__ ef_in,
                   const float* __restrict__ adj, float* __restrict__ merge_sum,
                   float* __restrict__ ssum) {
    const int N = 1024;
    const int bi = blockIdx.x, b = bi / N, i = bi % N;
    __shared__ float r1[256], r2[256];
    const float* er = ef_in + ((size_t)b * N + i) * N;
    const float* ar = adj + (size_t)i * N;
    const float* sr = sim + ((size_t)b * N + i) * N;
    float ms = 0.f, ss = 0.f;
    for (int j = threadIdx.x; j < N; j += 256) {
        float e = er[j] * ar[j];
        ms += e;
        ss += fabsf(sr[j] * e);
    }
    r1[threadIdx.x] = ms; r2[threadIdx.x] = ss;
    __syncthreads();
    for (int st = 128; st > 0; st >>= 1) {
        if (threadIdx.x < st) { r1[threadIdx.x] += r1[threadIdx.x + st];
                                r2[threadIdx.x] += r2[threadIdx.x + st]; }
        __syncthreads();
    }
    if (threadIdx.x == 0) { merge_sum[bi] = r1[0]; ssum[bi] = r2[0]; }
}

// ---------------------------------------------------------------------------
// Kernel 4: ef_out = l1norm(sim*edge*adj)*merge_sum + eye
//           ef_msk = ef_out with zeroed diagonal ; e2den = rowsum|ef_msk|
// ---------------------------------------------------------------------------
__global__ __launch_bounds__(256)
void efnew_kernel(const float* __restrict__ sim, const float* __restrict__ ef_in,
                  const float* __restrict__ adj, const float* __restrict__ merge_sum,
                  const float* __restrict__ ssum, float* __restrict__ ef_out,
                  float* __restrict__ ef_msk, float* __restrict__ e2den) {
    const int N = 1024;
    const int bi = blockIdx.x, b = bi / N, i = bi % N;
    const float inv = merge_sum[bi] / fmaxf(ssum[bi], NORM_EPS);
    const float* er = ef_in + ((size_t)b * N + i) * N;
    const float* ar = adj + (size_t)i * N;
    const float* sr = sim + ((size_t)b * N + i) * N;
    float* orow = ef_out + ((size_t)b * N + i) * N;
    float* mrow = ef_msk + ((size_t)b * N + i) * N;
    __shared__ float red[256];
    float acc = 0.f;
    for (int j = threadIdx.x; j < N; j += 256) {
        float v = sr[j] * er[j] * ar[j] * inv;
        float msk = (j == i) ? 0.f : v;
        orow[j] = v + ((j == i) ? 1.f : 0.f);
        mrow[j] = msk;
        acc += fabsf(msk);
    }
    red[threadIdx.x] = acc;
    __syncthreads();
    for (int st = 128; st > 0; st >>= 1) {
        if (threadIdx.x < st) red[threadIdx.x] += red[threadIdx.x + st];
        __syncthreads();
    }
    if (threadIdx.x == 0) e2den[bi] = red[0];
}

// ---------------------------------------------------------------------------
// Kernel 5 (WMMA): aggr[b,i,c] = (1/max(e2den,eps)) * sum_j ef_msk[b,i,j]*x[b,j,c]
// One wave per 16x16 output tile; K loop over j in steps of 32.
// ---------------------------------------------------------------------------
__global__ __launch_bounds__(128)
void aggr_kernel(const float* __restrict__ ef_msk, const float* __restrict__ e2den,
                 const float* __restrict__ x, float* __restrict__ aggr) {
    const int N = 1024, C = 12 * 32;
    const int waveg = (blockIdx.x * blockDim.x + threadIdx.x) >> 5;
    const int lane = threadIdx.x & 31;
    const int ct = waveg % (C / 16);
    const int it = (waveg / (C / 16)) % (N / 16);
    const int b = waveg / ((C / 16) * (N / 16));
    if (b >= 2) return;
    const int i0 = it * 16, c0 = ct * 16;
    const int m = lane & 15, ncol = lane & 15;
    const int khalf = (lane >> 4) << 4;
    const float* efrow = ef_msk + ((size_t)b * N + i0 + m) * N;
    v8f acc = {};
    for (int k0 = 0; k0 < N; k0 += 32) {
        v16h a;   // pure load + f16 convert, no masking math
#pragma unroll
        for (int e = 0; e < 16; ++e)
            a[e] = (_Float16)efrow[k0 + a_kidx(lane, e)];
        v16h bf;
#pragma unroll
        for (int e = 0; e < 16; ++e)
            bf[e] = (_Float16)x[((size_t)b * N + k0 + khalf + e) * C + c0 + ncol];
        acc = __builtin_amdgcn_wmma_f32_16x16x32_f16(false, a, false, bf,
                                                     (short)0, acc, false, false);
    }
    const int rowbase = (lane >> 4) << 3;
#pragma unroll
    for (int r = 0; r < 8; ++r) {
        int i = i0 + rowbase + r;
        float inv = 1.f / fmaxf(e2den[b * N + i], NORM_EPS);
        aggr[((size_t)b * N + i) * C + c0 + ncol] = acc[r] * inv;
    }
}

// ---------------------------------------------------------------------------
// Kernel 6: node MLP + residual + relu + LayerNorm. One wave32 per (b,n,t) row.
// ---------------------------------------------------------------------------
__global__ __launch_bounds__(256)
void node_kernel(const float* __restrict__ x, const float* __restrict__ aggr,
                 const float* __restrict__ w0, const float* __restrict__ w1,
                 const float* __restrict__ rw, const float* __restrict__ rb,
                 const float* __restrict__ lnw, const float* __restrict__ lnb,
                 float* __restrict__ xout) {
    const int ROWS = 2 * 1024 * 12;
    const int wavei = (blockIdx.x * blockDim.x + threadIdx.x) >> 5;
    const int lane = threadIdx.x & 31;
    const int nwaves = (gridDim.x * blockDim.x) >> 5;
    for (int row = wavei; row < ROWS; row += nwaves) {
        float xv = x[(size_t)row * 32 + lane];
        float av = aggr[(size_t)row * 32 + lane];
        float hA = 0.f, hB = 0.f;                 // h[lane], h[lane+32]
#pragma unroll
        for (int g = 0; g < 32; ++g) {
            float cx = __shfl(xv, g, 32);
            float ca = __shfl(av, g, 32);
            hA += w0[lane * 64 + g] * cx + w0[lane * 64 + 32 + g] * ca;
            hB += w0[(lane + 32) * 64 + g] * cx + w0[(lane + 32) * 64 + 32 + g] * ca;
        }
        hA = leaky(hA); hB = leaky(hB);
        float nodev = 0.f, xr = rb[lane];
#pragma unroll
        for (int g = 0; g < 32; ++g) {
            float h1 = __shfl(hA, g, 32);
            float h2 = __shfl(hB, g, 32);
            float cx = __shfl(xv, g, 32);
            nodev += w1[lane * 64 + g] * h1 + w1[lane * 64 + 32 + g] * h2;
            xr += rw[lane * 32 + g] * cx;
        }
        float y = fmaxf(xr + leaky(nodev), 0.f);
        float mu = y;
#pragma unroll
        for (int o = 16; o > 0; o >>= 1) mu += __shfl_xor(mu, o, 32);
        mu *= (1.f / 32.f);
        float d = y - mu;
        float var = d * d;
#pragma unroll
        for (int o = 16; o > 0; o >>= 1) var += __shfl_xor(var, o, 32);
        var *= (1.f / 32.f);
        xout[(size_t)row * 32 + lane] = d * rsqrtf(var + LN_EPS) * lnw[lane] + lnb[lane];
    }
}

// ---------------------------------------------------------------------------
extern "C" void kernel_launch(void* const* d_in, const int* in_sizes, int n_in,
                              void* d_out, int out_size, void* d_ws, size_t ws_size,
                              hipStream_t stream) {
    const int B = 2, N = 1024, T = 12, F = 32, L = 2;
    const float* x_in0  = (const float*)d_in[0];
    const float* edge   = (const float*)d_in[1];
    const float* adj    = (const float*)d_in[2];
    const float* tc_w   = (const float*)d_in[3];
    const float* tc_b   = (const float*)d_in[4];
    const float* e_w0   = (const float*)d_in[5];
    const float* e_w1   = (const float*)d_in[6];
    const float* e_wout = (const float*)d_in[7];
    const float* e_bout = (const float*)d_in[8];
    const float* n_w0   = (const float*)d_in[9];
    const float* n_w1   = (const float*)d_in[10];
    const float* r_w    = (const float*)d_in[11];
    const float* r_b    = (const float*)d_in[12];
    const float* ln_w   = (const float*)d_in[13];
    const float* ln_b   = (const float*)d_in[14];

    float* out_x  = (float*)d_out;                       // (B,N,T,F)
    float* out_ef = out_x + (size_t)B * N * T * F;       // (B,N,N)

    float* ws     = (float*)d_ws;                        // ~31.5 MB total
    float* nf     = ws;  ws += (size_t)B * N * F;
    float* sim    = ws;  ws += (size_t)B * N * N;
    float* ef_ws  = ws;  ws += (size_t)B * N * N;
    float* ef_msk = ws;  ws += (size_t)B * N * N;
    float* x_ws   = ws;  ws += (size_t)B * N * T * F;
    float* aggr   = ws;  ws += (size_t)B * N * T * F;
    float* msum   = ws;  ws += (size_t)B * N;
    float* ssum   = ws;  ws += (size_t)B * N;
    float* e2den  = ws;  ws += (size_t)B * N;

    for (int l = 0; l < L; ++l) {
        const float* xi  = (l == 0) ? x_in0 : x_ws;
        const float* efi = (l == 0) ? edge  : ef_ws;
        float* xo  = (l == 0) ? x_ws  : out_x;
        float* efo = (l == 0) ? ef_ws : out_ef;

        nf_kernel<<<(B * N * F + 255) / 256, 256, 0, stream>>>(
            xi, tc_w + l * T, tc_b + l, nf);
        sim_kernel<<<B * N, 128, 0, stream>>>(
            nf, e_w0 + (size_t)l * 64 * 32, e_w1 + (size_t)l * 32 * 64,
            e_wout + l * 32, e_bout + l, sim);
        rowsum_kernel<<<B * N, 256, 0, stream>>>(sim, efi, adj, msum, ssum);
        efnew_kernel<<<B * N, 256, 0, stream>>>(sim, efi, adj, msum, ssum,
                                                efo, ef_msk, e2den);
        // 2 * (1024/16) * (384/16) = 3072 waves -> 768 blocks of 128 threads
        aggr_kernel<<<768, 128, 0, stream>>>(ef_msk, e2den, xi, aggr);
        node_kernel<<<512, 256, 0, stream>>>(
            xi, aggr, n_w0 + (size_t)l * 64 * 64, n_w1 + (size_t)l * 32 * 64,
            r_w + (size_t)l * 32 * 32, r_b + l * 32, ln_w + l * 32, ln_b + l * 32, xo);
    }
}
